// Model_39676907881228
// MI455X (gfx1250) — matrix operand order repaired and checked by reading, not compiled, not run on hardware
//
#include <hip/hip_runtime.h>

// ---------------------------------------------------------------------------
// Flash-attention fwd, B=2,H=16,S=2048,D=64 fp32, MI455X (gfx1250, wave32).
// Compute-bound (34 GFLOP; K/V re-reads are L2-resident). fp32->f16 converted
// ONCE at LDS staging; WMMA operands are ds_load_b128 pairs per the CDNA5
// fragment tables:
//   A 16x32 f16: lane=M; halves h<8 -> K=lhalf*8+h, h>=8 -> K=16+lhalf*8+(h-8)
//   B 32x16 f16: lane=N; halves h<8 -> K=lhalf*8+h, h>=8 -> K=16+lhalf*8+(h-8)
// Each wave owns 32 Q rows (2 row tiles) so every B fragment read from LDS
// feeds two WMMAs; P row sums run on the matrix pipe (P x ones); only the
// row-max reduction uses ds_swizzle xor butterflies.
// ---------------------------------------------------------------------------

typedef __attribute__((ext_vector_type(16))) _Float16 v16h;
typedef __attribute__((ext_vector_type(8)))  _Float16 v8h;
typedef __attribute__((ext_vector_type(4)))  _Float16 v4h;
typedef __attribute__((ext_vector_type(8)))  float    v8f;

#define ATTN_B 2
#define ATTN_H 16
#define ATTN_S 2048
#define ATTN_D 64

#define QTILE  128     // q rows per block (4 waves x 32 rows)
#define KCHUNK 64      // keys per main-loop iteration
#define BLOCK  128
#define LDSTR  72      // halves per LDS row: 64 + 8 pad -> 144B rows
                       // (16B-aligned for b128, 36-bank stride = conflict-free)

// ds_swizzle_b32, group-of-32 mode, xor pattern: offset = and(0x1F) | xor<<10
template <int MASK>
static __device__ __forceinline__ float swz_xor(float v) {
    return __int_as_float(
        __builtin_amdgcn_ds_swizzle(__float_as_int(v), 0x1F | (MASK << 10)));
}

static __device__ __forceinline__ v16h cat8(v8h lo, v8h hi) {
    return __builtin_shufflevector(lo, hi, 0,1,2,3,4,5,6,7,8,9,10,11,12,13,14,15);
}

static __device__ __forceinline__ v4h cvt4(float4 f) {
    v4h r;
    r[0] = (_Float16)f.x; r[1] = (_Float16)f.y;
    r[2] = (_Float16)f.z; r[3] = (_Float16)f.w;
    return r;
}

// B operand: halves [0..8) and [16..24) per lane (caller adds lhalf*8)
static __device__ __forceinline__ v16h load_b_frag(const _Float16* p) {
    return cat8(*(const v8h*)(p), *(const v8h*)(p + 16));
}

#define WMMA_F16(A, B, C) \
    __builtin_amdgcn_wmma_f32_16x16x32_f16(false, (A), false, (B), (short)0, (C), false, false)

__global__ __launch_bounds__(BLOCK) void fa_fwd_kernel(
    const float* __restrict__ Q, const float* __restrict__ K,
    const float* __restrict__ V, const int* __restrict__ inv_scale_p,
    float* __restrict__ O)
{
    __shared__ __align__(16) _Float16 Kh [KCHUNK * LDSTR];            // [key][dim] f16
    __shared__ __align__(16) _Float16 Vth[ATTN_D * LDSTR];            // [dim][key] f16
    __shared__ __align__(16) _Float16 Ph [(BLOCK / 32) * 16 * LDSTR]; // per-wave P scratch

    const int tid   = threadIdx.x;
    const int wave  = tid >> 5;
    const int lane  = tid & 31;
    const int lhalf = lane >> 4;   // 0: lanes 0-15, 1: lanes 16-31
    const int l16   = lane & 15;

    const int qtiles = ATTN_S / QTILE;
    const int bh = blockIdx.x / qtiles;
    const int q0 = (blockIdx.x % qtiles) * QTILE + wave * 32;  // 32 rows per wave

    const size_t base  = (size_t)bh * ATTN_S * ATTN_D;
    const float  scale = 1.0f / (float)(*inv_scale_p);

    // ---- Q A-operand fragments: 2 row tiles x 2 K-splits, scale folded in ----
    v16h a[4];  // a[rt*2+0]: dims 0..31, a[rt*2+1]: dims 32..63
#pragma unroll
    for (int rt = 0; rt < 2; ++rt) {
        const float* qrow = Q + base + (size_t)(q0 + rt * 16 + l16) * ATTN_D;
#pragma unroll
        for (int i = 0; i < 8; ++i) {
            const int kk = ((i < 4) ? 0 : 16) + lhalf * 8 + 2 * (i & 3);
            float2 x = *(const float2*)(qrow + kk);
            a[rt*2+0][2*i]   = (_Float16)(x.x * scale);
            a[rt*2+0][2*i+1] = (_Float16)(x.y * scale);
            float2 y = *(const float2*)(qrow + 32 + kk);
            a[rt*2+1][2*i]   = (_Float16)(y.x * scale);
            a[rt*2+1][2*i+1] = (_Float16)(y.y * scale);
        }
    }

    // Constant all-ones B operand for matrix-pipe row sums (layout-invariant).
    v16h ones;
#pragma unroll
    for (int i = 0; i < 16; ++i) ones[i] = (_Float16)1.0f;

    // O(32x64): 2 row tiles x 4 N tiles of f32 C layout (lane=col, VGPR j=row)
    v8f ov[8] = {{}, {}, {}, {}, {}, {}, {}, {}};
    float mrow[16], lrow[16];
#pragma unroll
    for (int j = 0; j < 16; ++j) { mrow[j] = -3.0e38f; lrow[j] = 0.0f; }

    _Float16* pbuf = Ph + wave * (16 * LDSTR);

    const int skey  = tid >> 1;          // staging: this thread's K row
    const int shrow = (tid & 1) * 32;    // and which half of the row

    for (int kt = 0; kt < ATTN_S; kt += KCHUNK) {
        __syncthreads();  // previous iteration's LDS reads complete

        // ---- staging: fp32 -> f16 convert ONCE, shared by all 4 waves ----
        {
            const float4* ks = (const float4*)(K + base + (size_t)(kt + skey) * ATTN_D + shrow);
            v4h* kd = (v4h*)(Kh + skey * LDSTR + shrow);
#pragma unroll
            for (int i = 0; i < 8; ++i) kd[i] = cvt4(ks[i]);

            const float4* vs = (const float4*)(V + base + (size_t)kt * ATTN_D);
#pragma unroll
            for (int i = 0; i < 8; ++i) {
                const int f4 = i * BLOCK + tid;
                const float4 v = vs[f4];
                const int key = (f4 * 4) >> 6;
                const int dim = (f4 * 4) & 63;
                Vth[(dim + 0) * LDSTR + key] = (_Float16)v.x;
                Vth[(dim + 1) * LDSTR + key] = (_Float16)v.y;
                Vth[(dim + 2) * LDSTR + key] = (_Float16)v.z;
                Vth[(dim + 3) * LDSTR + key] = (_Float16)v.w;
            }
            if (kt + KCHUNK < ATTN_S) {  // global_prefetch_b8 of next chunk
                __builtin_prefetch(K + base + (size_t)(kt + KCHUNK) * ATTN_D + tid * 32, 0, 0);
                __builtin_prefetch(V + base + (size_t)(kt + KCHUNK) * ATTN_D + tid * 32, 0, 0);
            }
        }
        __syncthreads();  // staged data visible

        // ---- scores: 4 key tiles x 2 row tiles; each B fragment feeds 2 WMMAs ----
        v8f ct[8];  // ct[rt*4+t]
#pragma unroll
        for (int t = 0; t < 4; ++t) {
            const _Float16* krow = Kh + (t * 16 + l16) * LDSTR + lhalf * 8;
            v16h blo = load_b_frag(krow);        // dims  0..31
            v16h bhi = load_b_frag(krow + 32);   // dims 32..63
            v8f c0 = {}, c1 = {};
            c0 = WMMA_F16(a[0], blo, c0);
            c1 = WMMA_F16(a[2], blo, c1);
            c0 = WMMA_F16(a[1], bhi, c0);
            c1 = WMMA_F16(a[3], bhi, c1);
            ct[t]     = c0;
            ct[4 + t] = c1;
        }

        // ---- per row tile: online softmax, P store, pa fragments, row sums ----
        v16h pa[4];  // pa[rt*2+0] keys 0..31, pa[rt*2+1] keys 32..63
#pragma unroll
        for (int rt = 0; rt < 2; ++rt) {
            float alpha8[8];
#pragma unroll
            for (int j = 0; j < 8; ++j) {
                const int r = rt * 8 + j;
                float s = fmaxf(fmaxf(ct[rt*4+0][j], ct[rt*4+1][j]),
                                fmaxf(ct[rt*4+2][j], ct[rt*4+3][j]));
                s = fmaxf(s, swz_xor<1>(s));
                s = fmaxf(s, swz_xor<2>(s));
                s = fmaxf(s, swz_xor<4>(s));
                s = fmaxf(s, swz_xor<8>(s));
                const float mnew = fmaxf(mrow[r], s);
                const float p0 = __expf(ct[rt*4+0][j] - mnew);
                const float p1 = __expf(ct[rt*4+1][j] - mnew);
                const float p2 = __expf(ct[rt*4+2][j] - mnew);
                const float p3 = __expf(ct[rt*4+3][j] - mnew);
                const float alpha = __expf(mrow[r] - mnew);
                alpha8[j] = alpha;
                mrow[r] = mnew;
                ov[rt*4+0][j] *= alpha; ov[rt*4+1][j] *= alpha;
                ov[rt*4+2][j] *= alpha; ov[rt*4+3][j] *= alpha;
                // P (C layout: lane = key column) -> per-wave LDS scratch as f16
                _Float16* pr = pbuf + (lhalf * 8 + j) * LDSTR + l16;
                pr[0]  = (_Float16)p0;
                pr[16] = (_Float16)p1;
                pr[32] = (_Float16)p2;
                pr[48] = (_Float16)p3;
            }
            // P back as A-operand (DS ops of one wave are in-order, so these
            // loads complete before the next row tile's stores overwrite pbuf)
            const _Float16* prow = pbuf + l16 * LDSTR;  // lane = row M
            pa[rt*2+0] = cat8(*(const v8h*)(prow +      lhalf * 8),
                              *(const v8h*)(prow + 16 + lhalf * 8));
            pa[rt*2+1] = cat8(*(const v8h*)(prow + 32 + lhalf * 8),
                              *(const v8h*)(prow + 48 + lhalf * 8));
            // row sums on the matrix pipe: every lane gets its 8 rows' sums
            v8f rsum = {};
            rsum = WMMA_F16(pa[rt*2+0], ones, rsum);
            rsum = WMMA_F16(pa[rt*2+1], ones, rsum);
#pragma unroll
            for (int j = 0; j < 8; ++j) {
                const int r = rt * 8 + j;
                lrow[r] = lrow[r] * alpha8[j] + rsum[j];
            }
        }

        // ---- O += P . V: each V fragment feeds both row tiles ----
#pragma unroll
        for (int nt = 0; nt < 4; ++nt) {
            const _Float16* vrow = Vth + (nt * 16 + l16) * LDSTR + lhalf * 8;
            v16h bv0 = load_b_frag(vrow);        // keys  0..31
            v16h bv1 = load_b_frag(vrow + 32);   // keys 32..63
            ov[nt]     = WMMA_F16(pa[0], bv0, ov[nt]);
            ov[4 + nt] = WMMA_F16(pa[2], bv0, ov[4 + nt]);
            ov[nt]     = WMMA_F16(pa[1], bv1, ov[nt]);
            ov[4 + nt] = WMMA_F16(pa[3], bv1, ov[4 + nt]);
        }
    }

    // ---- epilogue: divide by running sum, store fp32 ----
#pragma unroll
    for (int rt = 0; rt < 2; ++rt) {
#pragma unroll
        for (int j = 0; j < 8; ++j) {
            const float inv_l = 1.0f / lrow[rt * 8 + j];
            float* orow = O + base + (size_t)(q0 + rt * 16 + lhalf * 8 + j) * ATTN_D + l16;
            orow[0]  = ov[rt*4+0][j] * inv_l;
            orow[16] = ov[rt*4+1][j] * inv_l;
            orow[32] = ov[rt*4+2][j] * inv_l;
            orow[48] = ov[rt*4+3][j] * inv_l;
        }
    }
}

extern "C" void kernel_launch(void* const* d_in, const int* in_sizes, int n_in,
                              void* d_out, int out_size, void* d_ws, size_t ws_size,
                              hipStream_t stream) {
    (void)in_sizes; (void)n_in; (void)out_size; (void)d_ws; (void)ws_size;
    const float* Q   = (const float*)d_in[0];
    const float* Kp  = (const float*)d_in[1];
    const float* Vp  = (const float*)d_in[2];
    const int*   inv = (const int*)d_in[3];   // inv_scale_factor (python int)
    // d_in[4] = dropout_p == 0 (deterministic reference) -> ignored
    float* out = (float*)d_out;

    dim3 grid(ATTN_B * ATTN_H * (ATTN_S / QTILE));
    dim3 block(BLOCK);
    fa_fwd_kernel<<<grid, block, 0, stream>>>(Q, Kp, Vp, inv, out);
}
